// NCM_61349312856216
// MI455X (gfx1250) — matrix operand order repaired and checked by reading, not compiled, not run on hardware
//
#include <hip/hip_runtime.h>

// MI455X / gfx1250: wave32, WMMA 16x16x32 f16->f32, 320KB LDS per WGP.

typedef _Float16 v16h __attribute__((ext_vector_type(16)));
typedef _Float16 v8h  __attribute__((ext_vector_type(8)));
typedef float    v8f  __attribute__((ext_vector_type(8)));

#define M_SAMP 32768
#define NN 64          // nodes
#define HH 32          // hidden

// ---------------------------------------------------------------------------
// Prep: pack binary adjacency column A[m, :, node] into a 64-bit parent mask.
// Thread t -> (m = t>>6, node = t&63). Consecutive threads hit consecutive
// 'node' addresses -> fully coalesced; A (512MB) is read exactly once.
// ---------------------------------------------------------------------------
__global__ __launch_bounds__(256) void ncm_pack_bits(
    const float* __restrict__ A, unsigned long long* __restrict__ bits)
{
  const int t = blockIdx.x * blockDim.x + threadIdx.x;   // 0 .. M*64-1
  const int m = t >> 6;
  const int node = t & 63;
  const float* base = A + ((size_t)m << 12) + node;      // A[m, i, node], stride 64 floats in i
  unsigned long long b = 0ull;
#pragma unroll
  for (int i = 0; i < 64; ++i)
    b |= (unsigned long long)(base[(size_t)i << 6] != 0.0f) << i;
  bits[t] = b;
}

// ---------------------------------------------------------------------------
// Main: one wave32 per sample. All node MLP weights live in LDS as f16
// (W1 first 64 features) + f32 (b1, W1[:,64], W2, b2). Per sequential step:
//   - build x (masked outputs) as f16 in a wave-private LDS buffer
//   - 4x v_wmma_f32_16x16x32_f16: rows [0:16),[16:32) x K-chunks [0:32),[32:64)
//     with B = x-chunk replicated across the 16 columns, C preloaded with
//     b1 + z*W1[:,64]
//   - leaky_relu, dot with W2 (per-lane FMA + wave32 xor-shuffle reduction)
// ---------------------------------------------------------------------------
__global__ __launch_bounds__(128) void ncm_main(
    const float* __restrict__ Z,
    const unsigned long long* __restrict__ Abits,
    const float* __restrict__ A,
    const float* __restrict__ u,
    const float* __restrict__ W1,      // (N, H, N+1)
    const float* __restrict__ b1,      // (N, H)
    const float* __restrict__ W2,      // (N, 1, H)
    const float* __restrict__ b2,      // (N, 1)
    const int*   __restrict__ order,   // (M, N)
    const int*   __restrict__ dov,     // (M,)
    float* __restrict__ out,           // (M, N)
    int useAbits, int samplesPerWave)
{
  __shared__ __align__(16) _Float16 W1s[NN * HH * 64];   // 256 KB, [node][h][k<64]
  __shared__ __align__(16) float    b1s[NN * HH];        // 8 KB
  __shared__ __align__(16) float    w1zs[NN * HH];       // 8 KB (W1[:, :, 64])
  __shared__ __align__(16) float    w2s[NN * HH];        // 8 KB
  __shared__ __align__(16) float    b2s[NN];
  __shared__ __align__(32) _Float16 xh[4][64];           // wave-private x staging

  const int tid  = threadIdx.x;
  const int lane = tid & 31;
  const int wave = tid >> 5;

  // --- fill LDS weight cache (f32 -> f16 on the fly) ---
  for (int i = tid; i < NN * HH; i += blockDim.x) {
    const float* wr = W1 + (size_t)i * 65;
    _Float16* dr = &W1s[(size_t)i * 64];
#pragma unroll 8
    for (int k = 0; k < 64; ++k) dr[k] = (_Float16)wr[k];
    w1zs[i] = wr[64];
    b1s[i]  = b1[i];
    w2s[i]  = W2[i];
  }
  if (tid < NN) b2s[tid] = b2[tid];
  __syncthreads();

  const int wavesPerBlk = blockDim.x >> 5;
  const int gwave = blockIdx.x * wavesPerBlk + wave;
  _Float16* xw = xh[wave];

  const int hi16  = lane >> 4;        // 0 | 1 (lane half)
  const int rb    = hi16 << 3;        // C/D row offset: 0 | 8
  const int khalf = hi16 << 4;        // unused; A and B both use ksub
  const int ksub  = hi16 << 3;        // A K offset:     0 | 8
  const int r0    = lane & 15;        // A row within 16-block

  for (int si = 0; si < samplesPerWave; ++si) {
    const int m = gwave * samplesPerWave + si;
    const int* ord = order + ((size_t)m << 6);
    const int dv   = dov[m];
    const float uv = u[m];
    // outputs[] striped across lanes: lane holds indices {lane, lane+32}
    float o0 = (dv == lane)      ? uv : 0.0f;
    float o1 = (dv == lane + 32) ? uv : 0.0f;

    for (int t = 0; t < NN; ++t) {
      const int node = ord[t];

      float x0, x1;
      if (useAbits) {
        const unsigned long long bits = Abits[((size_t)m << 6) + node];
        x0 = ((bits >> lane) & 1ull)        ? o0 : 0.0f;
        x1 = ((bits >> (lane + 32)) & 1ull) ? o1 : 0.0f;
      } else {
        const float* acol = A + ((size_t)m << 12) + node;
        x0 = (acol[(size_t)lane << 6]        != 0.0f) ? o0 : 0.0f;
        x1 = (acol[(size_t)(lane + 32) << 6] != 0.0f) ? o1 : 0.0f;
      }
      xw[lane]      = (_Float16)x0;
      xw[lane + 32] = (_Float16)x1;
      __builtin_amdgcn_wave_barrier();   // keep LDS stores before the B reads

      // B operands: x chunk replicated over all 16 columns.
      // 32x16 f16 B layout: lanes 0-15 hold K=[0,8)+[16,24), lanes 16-31 K=[8,16)+[24,32).
      v16h B0 = __builtin_shufflevector(
          *(const v8h*)(xw + ksub),      *(const v8h*)(xw + ksub + 16),
          0,1,2,3,4,5,6,7,8,9,10,11,12,13,14,15);
      v16h B1 = __builtin_shufflevector(
          *(const v8h*)(xw + 32 + ksub), *(const v8h*)(xw + 32 + ksub + 16),
          0,1,2,3,4,5,6,7,8,9,10,11,12,13,14,15);

      // C init = b1 + z * W1[:,64]  (folds the 65th feature into the WMMA)
      const float z = Z[((size_t)m << 6) + node];
      const int base = node * HH + rb;
      v8f c0, c1;
#pragma unroll
      for (int v = 0; v < 8; ++v) {
        c0[v] = __builtin_fmaf(z, w1zs[base + v],      b1s[base + v]);
        c1[v] = __builtin_fmaf(z, w1zs[base + 16 + v], b1s[base + 16 + v]);
      }

      // A operands: 16x32 f16 tiles of W1[node] from LDS (two b128 per tile).
      const _Float16* wrow = W1s + (size_t)node * (HH * 64);
      const _Float16* pa0 = wrow + (size_t)r0 * 64 + ksub;          // rows 0..15
      const _Float16* pa1 = wrow + (size_t)(16 + r0) * 64 + ksub;   // rows 16..31
      v16h A00 = __builtin_shufflevector(*(const v8h*)pa0,        *(const v8h*)(pa0 + 16),
          0,1,2,3,4,5,6,7,8,9,10,11,12,13,14,15);
      v16h A01 = __builtin_shufflevector(*(const v8h*)(pa0 + 32), *(const v8h*)(pa0 + 48),
          0,1,2,3,4,5,6,7,8,9,10,11,12,13,14,15);
      v16h A10 = __builtin_shufflevector(*(const v8h*)pa1,        *(const v8h*)(pa1 + 16),
          0,1,2,3,4,5,6,7,8,9,10,11,12,13,14,15);
      v16h A11 = __builtin_shufflevector(*(const v8h*)(pa1 + 32), *(const v8h*)(pa1 + 48),
          0,1,2,3,4,5,6,7,8,9,10,11,12,13,14,15);

      v8f acc0 = __builtin_amdgcn_wmma_f32_16x16x32_f16(false, A00, false, B0,
                                                        (short)0, c0, false, false);
      acc0     = __builtin_amdgcn_wmma_f32_16x16x32_f16(false, A01, false, B1,
                                                        (short)0, acc0, false, false);
      v8f acc1 = __builtin_amdgcn_wmma_f32_16x16x32_f16(false, A10, false, B0,
                                                        (short)0, c1, false, false);
      acc1     = __builtin_amdgcn_wmma_f32_16x16x32_f16(false, A11, false, B1,
                                                        (short)0, acc1, false, false);
      __builtin_amdgcn_wave_barrier();

      // leaky_relu (slope 0.01) + dot with W2. All 16 B-columns are identical,
      // so the wave-wide sum is 16x the true dot product.
      float s = 0.0f;
#pragma unroll
      for (int v = 0; v < 8; ++v) {
        const float p0 = acc0[v], p1 = acc1[v];
        const float h0 = fmaxf(p0, 0.01f * p0);
        const float h1 = fmaxf(p1, 0.01f * p1);
        s = __builtin_fmaf(h0, w2s[base + v],      s);
        s = __builtin_fmaf(h1, w2s[base + 16 + v], s);
      }
#pragma unroll
      for (int off = 16; off; off >>= 1) s += __shfl_xor(s, off, 32);
      const float y = s * 0.0625f + b2s[node];

      if (dv != node) {                 // intervened node keeps its value
        if (lane == (node & 31)) {
          if (node < 32) o0 = y; else o1 = y;
        }
      }
    }

    out[((size_t)m << 6) + lane]      = o0;
    out[((size_t)m << 6) + 32 + lane] = o1;
  }
}

// ---------------------------------------------------------------------------
extern "C" void kernel_launch(void* const* d_in, const int* in_sizes, int n_in,
                              void* d_out, int out_size, void* d_ws, size_t ws_size,
                              hipStream_t stream) {
  (void)in_sizes; (void)n_in; (void)out_size;
  const float* Z     = (const float*)d_in[0];
  const float* A     = (const float*)d_in[1];
  const float* u     = (const float*)d_in[2];
  const float* W1    = (const float*)d_in[3];
  const float* b1    = (const float*)d_in[4];
  const float* W2    = (const float*)d_in[5];
  const float* b2    = (const float*)d_in[6];
  const int*   order = (const int*)d_in[7];
  const int*   dov   = (const int*)d_in[8];
  float* out = (float*)d_out;

  unsigned long long* Abits = (unsigned long long*)d_ws;
  const size_t bitsBytes = (size_t)M_SAMP * NN * sizeof(unsigned long long); // 16 MB
  const int useAbits = (ws_size >= bitsBytes) ? 1 : 0;

  if (useAbits) {
    const int total = M_SAMP * NN;                 // 2M threads
    ncm_pack_bits<<<total / 256, 256, 0, stream>>>(A, Abits);
  }

  const int blocks = 512, threads = 128;           // 2048 waves -> 16 samples/wave
  const int samplesPerWave = M_SAMP / (blocks * (threads / 32));
  ncm_main<<<blocks, threads, 0, stream>>>(Z, Abits, A, u, W1, b1, W2, b2,
                                           order, dov, out, useAbits, samplesPerWave);
}
